// CrossAttentionBlock_11974368821817
// MI455X (gfx1250) — matrix operand order repaired and checked by reading, not compiled, not run on hardware
//
#include <hip/hip_runtime.h>
#include <hip/hip_bf16.h>

typedef __attribute__((ext_vector_type(16))) _Float16 v16h;
typedef __attribute__((ext_vector_type(8)))  _Float16 v8h;
typedef __attribute__((ext_vector_type(8)))  float    v8f;

constexpr int Bc = 2, TQc = 2048, TKc = 2048, Cc = 1024, Hc = 16, Dc = 64;

enum { EPI_F32 = 0, EPI_RESID_F32 = 1, EPI_GELU_F16 = 2 };

// ---------------------------------------------------------------- helpers ---
__device__ __forceinline__ v16h ld16(const _Float16* p0, const _Float16* p1) {
  v8h a = *(const v8h*)p0;
  v8h b = *(const v8h*)p1;
  v16h r;
#pragma unroll
  for (int i = 0; i < 8; ++i) { r[i] = a[i]; r[8 + i] = b[i]; }
  return r;
}

__device__ __forceinline__ v8f wmma_f32_f16(v16h a, v16h b, v8f c) {
  // 8 args: (neg_a, A, neg_b, B, c_mod, C, reuse_a, reuse_b)
  return __builtin_amdgcn_wmma_f32_16x16x32_f16(false, a, false, b, (short)0, c,
                                                false, false);
}

// DPP16 ROW_XMASK butterfly exchange within 16-lane rows (no LDS, no dscnt).
template <int MASK>
__device__ __forceinline__ float dpp_xor16(float x) {
  int v = __builtin_amdgcn_update_dpp(0, __builtin_bit_cast(int, x),
                                      0x160 | MASK, 0xf, 0xf, true);
  return __builtin_bit_cast(float, v);
}
__device__ __forceinline__ float red16_max(float x) {
  x = fmaxf(x, dpp_xor16<1>(x));
  x = fmaxf(x, dpp_xor16<2>(x));
  x = fmaxf(x, dpp_xor16<4>(x));
  x = fmaxf(x, dpp_xor16<8>(x));
  return x;
}
__device__ __forceinline__ float red16_min(float x) {
  x = fminf(x, dpp_xor16<1>(x));
  x = fminf(x, dpp_xor16<2>(x));
  x = fminf(x, dpp_xor16<4>(x));
  x = fminf(x, dpp_xor16<8>(x));
  return x;
}
__device__ __forceinline__ float red16_sum(float x) {
  x += dpp_xor16<1>(x);
  x += dpp_xor16<2>(x);
  x += dpp_xor16<4>(x);
  x += dpp_xor16<8>(x);
  return x;
}

// A-fragment 16x32 f16 (rows striped in-lane): lane<16 holds K {0..7,16..23},
// lane>=16 holds K {8..15,24..31}; row = lane&15.
__device__ __forceinline__ v16h load_A(const _Float16* base, int ld, int row0,
                                       int k0, int lane) {
  int m = lane & 15, hs = (lane >> 4) & 1;
  const _Float16* p = base + (size_t)(row0 + m) * ld + k0 + hs * 8;
  return ld16(p, p + 16);
}

// B-fragment 32x16 f16: column n = lane&15 (== row n of W, contiguous K);
// lane<16 holds K {0..7,16..23}, lane>=16 holds K {8..15,24..31}.
__device__ __forceinline__ v16h load_B(const _Float16* base, int ld, int n0,
                                       int k0, int lane) {
  int n = lane & 15, hs = (lane >> 4) & 1;
  const _Float16* p = base + (size_t)(n0 + n) * ld + k0 + hs * 8;
  return ld16(p, p + 16);
}

// ------------------------------------------------------------ cast / LN -----
__global__ void cast_f32_f16_kernel(const float* __restrict__ in,
                                    _Float16* __restrict__ out, int n) {
  int i = blockIdx.x * blockDim.x + threadIdx.x;
  if (i < n) out[i] = (_Float16)in[i];
}

__global__ __launch_bounds__(256) void ln_cast_kernel(
    const float* __restrict__ in, const float* __restrict__ w,
    _Float16* __restrict__ out, int C_) {
  int row = blockIdx.x;
  const float* p = in + (size_t)row * C_;
  __shared__ float smem[16];
  __shared__ float mu_s, rstd_s;
  float s = 0.f, s2 = 0.f;
  for (int i = threadIdx.x; i < C_; i += 256) {
    float v = p[i];
    s += v;
    s2 = fmaf(v, v, s2);
  }
#pragma unroll
  for (int off = 16; off >= 1; off >>= 1) {
    s += __shfl_xor(s, off);
    s2 += __shfl_xor(s2, off);
  }
  int wid = threadIdx.x >> 5;
  if ((threadIdx.x & 31) == 0) { smem[wid] = s; smem[8 + wid] = s2; }
  __syncthreads();
  if (threadIdx.x == 0) {
    float ts = 0.f, ts2 = 0.f;
    for (int i = 0; i < 8; ++i) { ts += smem[i]; ts2 += smem[8 + i]; }
    float mu = ts / C_;
    float var = ts2 / C_ - mu * mu;
    mu_s = mu;
    rstd_s = rsqrtf(var + 1e-5f);
  }
  __syncthreads();
  float mu = mu_s, rstd = rstd_s;
  for (int i = threadIdx.x; i < C_; i += 256)
    out[(size_t)row * C_ + i] = (_Float16)((p[i] - mu) * rstd * w[i]);
}

// ------------------------------------------------------------ WMMA GEMM -----
// out[M,N] = epilogue(A[M,K] @ W[N,K]^T).  One wave -> 16x64 tile.
// Fragments are double-buffered: chunk k+32 is loaded before chunk k's WMMAs
// issue, so global-load latency overlaps matrix math.
template <int EPI>
__global__ __launch_bounds__(256) void gemm_wmma_kernel(
    const _Float16* __restrict__ A, const _Float16* __restrict__ W,
    const float* __restrict__ resid, float* __restrict__ out_f32,
    _Float16* __restrict__ out_f16, int M, int N, int K) {
  int wave = threadIdx.x >> 5;
  int lane = threadIdx.x & 31;
  int tiles_n = N >> 6;
  int tile = blockIdx.x * 8 + wave;
  if (tile >= (M >> 4) * tiles_n) return;
  int tm = tile / tiles_n, tn = tile - tm * tiles_n;
  int row0 = tm << 4;
  int m = lane & 15, hs = (lane >> 4) & 1;

  v8f acc[4] = {};
  const _Float16* arow = A + (size_t)(row0 + m) * K + hs * 8;
  const _Float16* wrow = W + (size_t)((tn << 6) + m) * K + hs * 8;

  v16h a = ld16(arow, arow + 16);
  v16h b[4];
#pragma unroll
  for (int j = 0; j < 4; ++j) {
    const _Float16* wp = wrow + (size_t)j * 16 * K;
    b[j] = ld16(wp, wp + 16);
  }

  for (int k0 = 0; k0 < K - 32; k0 += 32) {
    __builtin_prefetch(arow + k0 + 288, 0, 1);  // global_prefetch_b8
    // issue next chunk's loads first ...
    v16h an = ld16(arow + k0 + 32, arow + k0 + 48);
    v16h bn[4];
#pragma unroll
    for (int j = 0; j < 4; ++j) {
      const _Float16* wp = wrow + (size_t)j * 16 * K + k0 + 32;
      bn[j] = ld16(wp, wp + 16);
    }
    // ... then this chunk's WMMAs (loads in flight underneath)
#pragma unroll
    for (int j = 0; j < 4; ++j) acc[j] = wmma_f32_f16(a, b[j], acc[j]);
    a = an;
#pragma unroll
    for (int j = 0; j < 4; ++j) b[j] = bn[j];
  }
#pragma unroll
  for (int j = 0; j < 4; ++j) acc[j] = wmma_f32_f16(a, b[j], acc[j]);

  int col0 = (tn << 6) + m;
#pragma unroll
  for (int j = 0; j < 4; ++j) {
    int col = col0 + j * 16;
#pragma unroll
    for (int r = 0; r < 8; ++r) {
      size_t idx = (size_t)(row0 + r + hs * 8) * N + col;
      float v = acc[j][r];
      if (EPI == EPI_RESID_F32) {
        out_f32[idx] = v + resid[idx];
      } else if (EPI == EPI_GELU_F16) {
        out_f16[idx] = (_Float16)(0.5f * v * (1.0f + erff(v * 0.70710678f)));
      } else {
        out_f32[idx] = v;
      }
    }
  }
}

// -------------------------------------------------------------- RoPE --------
// `scale` folds the 1/sqrt(D) softmax scale into q (exact in f16: 2^-3).
__global__ void rope_kernel(const float* __restrict__ src, int src_stride,
                            const float* __restrict__ times,
                            const float* __restrict__ freqs,
                            _Float16* __restrict__ dst, int T, float scale) {
  int idx = blockIdx.x * blockDim.x + threadIdx.x;
  if (idx >= Bc * Hc * T * (Dc / 2)) return;
  int d2 = idx & 31;          // D/2 = 32
  int h = (idx >> 5) & 15;    // H  = 16
  int bt = idx >> 9;          // b*T + t
  int b = bt / T, t = bt - b * T;
  float ang = times[bt] * freqs[d2];
  float cs = cosf(ang), sn = sinf(ang);
  const float* s = src + (size_t)bt * src_stride + h * Dc + d2 * 2;
  float x1 = s[0], x2 = s[1];
  _Float16* d = dst + ((size_t)(b * Hc + h) * T + t) * Dc + d2 * 2;
  d[0] = (_Float16)((x1 * cs - x2 * sn) * scale);
  d[1] = (_Float16)((x1 * sn + x2 * cs) * scale);
}

// v slice of kv -> transposed f16 [B,H,D,TK]
__global__ void cast_v_kernel(const float* __restrict__ kvraw,
                              _Float16* __restrict__ vt) {
  int idx = blockIdx.x * blockDim.x + threadIdx.x;
  if (idx >= Bc * TKc * Cc) return;
  int d = idx & 63;
  int h = (idx >> 6) & 15;
  int bt = idx >> 10;
  int b = bt >> 11, t = bt & (TKc - 1);
  float v = kvraw[(size_t)bt * (2 * Cc) + Cc + h * Dc + d];
  vt[((size_t)(b * Hc + h) * Dc + d) * TKc + t] = (_Float16)v;
}

// ------------------------------------------------------- flash attention ----
__global__ __launch_bounds__(128) void flash_attn_kernel(
    const _Float16* __restrict__ qh, const _Float16* __restrict__ kh,
    const _Float16* __restrict__ vt, const float* __restrict__ yt,
    const float* __restrict__ xt, const int* __restrict__ min_dist,
    _Float16* __restrict__ oh) {
  __shared__ _Float16 pbuf_all[4][16 * 32];
  int lane = threadIdx.x & 31;
  int wave = threadIdx.x >> 5;
  _Float16* pbuf = pbuf_all[wave];
  int q0 = (blockIdx.x * 4 + wave) * 16;
  int h = blockIdx.y;
  int b = blockIdx.z;
  size_t bh = (size_t)(b * Hc + h);
  const _Float16* qbase = qh + bh * TQc * Dc;
  const _Float16* kbase = kh + bh * TKc * Dc;
  const _Float16* vbase = vt + bh * Dc * TKc;
  int m = lane & 15, hs = (lane >> 4) & 1;

  v16h aq0 = load_A(qbase, Dc, q0, 0, lane);
  v16h aq1 = load_A(qbase, Dc, q0, 32, lane);

  float md = (float)min_dist[0];
  float rt[8], mi[8], li[8];
#pragma unroll
  for (int r = 0; r < 8; ++r) {
    rt[r] = yt[b * TQc + q0 + r + hs * 8] - md;
    mi[r] = -INFINITY;
    li[r] = 0.f;
  }
  float rtmax = rt[0];
#pragma unroll
  for (int r = 1; r < 8; ++r) rtmax = fmaxf(rtmax, rt[r]);
  rtmax = red16_max(rtmax);
  rtmax = fmaxf(rtmax, __shfl_xor(rtmax, 16));  // once, outside hot loop

  v8f o[4] = {};
  v8f zero = {};

  for (int kb = 0; kb < TKc; kb += 32) {
    float kt0 = xt[b * TKc + kb + m];
    float kt1 = xt[b * TKc + kb + 16 + m];
    float ktmin = red16_min(fminf(kt0, kt1));
    // wave-uniform skip through SALU so EXEC stays all-ones for WMMA
    if (__builtin_amdgcn_readfirstlane(rtmax < ktmin ? 1 : 0)) continue;

    __builtin_prefetch(kbase + (size_t)(kb + 32 + m) * Dc, 0, 1);
    __builtin_prefetch(vbase + (size_t)m * TKc + kb + 32, 0, 1);

    // K fragments (needed immediately by score WMMAs) ...
    v16h bk0 = load_B(kbase, Dc, kb, 0, lane);
    v16h bk1 = load_B(kbase, Dc, kb, 32, lane);
    v16h bk2 = load_B(kbase, Dc, kb + 16, 0, lane);
    v16h bk3 = load_B(kbase, Dc, kb + 16, 32, lane);
    // ... V fragments issued now, consumed only after the softmax section:
    // their latency hides under ~100 VALU ops.
    v16h bv0 = load_B(vbase, TKc, 0, kb, lane);
    v16h bv1 = load_B(vbase, TKc, 16, kb, lane);
    v16h bv2 = load_B(vbase, TKc, 32, kb, lane);
    v16h bv3 = load_B(vbase, TKc, 48, kb, lane);

    // S = (q/sqrt(D)) @ K^T (scale pre-folded into q by rope)
    v8f s0 = wmma_f32_f16(aq0, bk0, zero);
    s0 = wmma_f32_f16(aq1, bk1, s0);
    v8f s1 = wmma_f32_f16(aq0, bk2, zero);
    s1 = wmma_f32_f16(aq1, bk3, s1);

    float f[8];
#pragma unroll
    for (int r = 0; r < 8; ++r) {
      float a0 = (rt[r] >= kt0) ? s0[r] : -INFINITY;
      float a1 = (rt[r] >= kt1) ? s1[r] : -INFINITY;
      float mnew = fmaxf(mi[r], red16_max(fmaxf(a0, a1)));
      float fac = (mnew > -1e30f) ? __expf(mi[r] - mnew) : 1.0f;
      float p0 = (a0 > -1e30f) ? __expf(a0 - mnew) : 0.f;
      float p1 = (a1 > -1e30f) ? __expf(a1 - mnew) : 0.f;
      li[r] = li[r] * fac + red16_sum(p0 + p1);
      mi[r] = mnew;
      f[r] = fac;
      // C-layout -> LDS (row-major 16x32) for A-layout reload
      pbuf[(r + hs * 8) * 32 + m] = (_Float16)p0;
      pbuf[(r + hs * 8) * 32 + 16 + m] = (_Float16)p1;
    }
#pragma unroll
    for (int j = 0; j < 4; ++j)
#pragma unroll
      for (int r = 0; r < 8; ++r) o[j][r] *= f[r];

    asm volatile("s_wait_dscnt 0" ::: "memory");
    const _Float16* prow = pbuf + m * 32 + hs * 8;
    v16h ap = ld16(prow, prow + 16);  // P as A-fragment (16x32)

    // O += P @ V (V fragments already resident)
    o[0] = wmma_f32_f16(ap, bv0, o[0]);
    o[1] = wmma_f32_f16(ap, bv1, o[1]);
    o[2] = wmma_f32_f16(ap, bv2, o[2]);
    o[3] = wmma_f32_f16(ap, bv3, o[3]);
  }

#pragma unroll
  for (int j = 0; j < 4; ++j) {
#pragma unroll
    for (int r = 0; r < 8; ++r) {
      float inv = (li[r] > 0.f) ? 1.0f / li[r] : 0.f;
      size_t row = (size_t)(b * TQc + q0 + r + hs * 8);
      oh[row * Cc + h * Dc + j * 16 + m] = (_Float16)(o[j][r] * inv);
    }
  }
}

// ------------------------------------------------------------- launcher -----
extern "C" void kernel_launch(void* const* d_in, const int* in_sizes, int n_in,
                              void* d_out, int out_size, void* d_ws,
                              size_t ws_size, hipStream_t stream) {
  (void)in_sizes; (void)n_in; (void)out_size; (void)ws_size;
  const float* y = (const float*)d_in[0];
  const float* x = (const float*)d_in[1];
  const float* y_t = (const float*)d_in[2];
  const float* x_t = (const float*)d_in[3];
  const float* ln1w = (const float*)d_in[4];
  const float* ln3w = (const float*)d_in[5];
  const float* ln2w = (const float*)d_in[6];
  const float* Wq = (const float*)d_in[7];
  const float* Wkv = (const float*)d_in[8];
  const float* Wproj = (const float*)d_in[9];
  const float* Wfc = (const float*)d_in[10];
  const float* Wmlp = (const float*)d_in[11];
  const float* freqs = (const float*)d_in[12];
  const int* mind = (const int*)d_in[13];

  char* ws = (char*)d_ws;
  const size_t MB = 1024 * 1024;
  _Float16* wqh = (_Float16*)(ws + 0 * MB);      // 2 MB
  _Float16* wkvh = (_Float16*)(ws + 2 * MB);     // 4 MB
  _Float16* wprojh = (_Float16*)(ws + 6 * MB);   // 2 MB
  _Float16* wfch = (_Float16*)(ws + 8 * MB);     // 2 MB
  _Float16* wmlph = (_Float16*)(ws + 10 * MB);   // 2 MB
  _Float16* qnh = (_Float16*)(ws + 12 * MB);     // 8 MB (reused as LN2 out)
  _Float16* knh = (_Float16*)(ws + 20 * MB);     // 8 MB (reused as FC out)
  float* qraw = (float*)(ws + 28 * MB);          // 16 MB (reused as y1)
  float* kvraw = (float*)(ws + 44 * MB);         // 32 MB
  _Float16* qh = (_Float16*)(ws + 76 * MB);      // 8 MB
  _Float16* kh = (_Float16*)(ws + 84 * MB);      // 8 MB
  _Float16* vt = (_Float16*)(ws + 92 * MB);      // 8 MB
  _Float16* oh = (_Float16*)(ws + 100 * MB);     // 8 MB

  const int rows = Bc * TQc;  // 4096
  const int nC = Cc * Cc;     // 1M
  // weights -> f16
  cast_f32_f16_kernel<<<(nC + 255) / 256, 256, 0, stream>>>(Wq, wqh, nC);
  cast_f32_f16_kernel<<<(2 * nC + 255) / 256, 256, 0, stream>>>(Wkv, wkvh, 2 * nC);
  cast_f32_f16_kernel<<<(nC + 255) / 256, 256, 0, stream>>>(Wproj, wprojh, nC);
  cast_f32_f16_kernel<<<(nC + 255) / 256, 256, 0, stream>>>(Wfc, wfch, nC);
  cast_f32_f16_kernel<<<(nC + 255) / 256, 256, 0, stream>>>(Wmlp, wmlph, nC);
  // layernorms (+cast)
  ln_cast_kernel<<<rows, 256, 0, stream>>>(y, ln1w, qnh, Cc);
  ln_cast_kernel<<<rows, 256, 0, stream>>>(x, ln3w, knh, Cc);
  // q / kv projections
  gemm_wmma_kernel<EPI_F32><<<512, 256, 0, stream>>>(qnh, wqh, nullptr, qraw,
                                                     nullptr, rows, Cc, Cc);
  gemm_wmma_kernel<EPI_F32><<<1024, 256, 0, stream>>>(knh, wkvh, nullptr, kvraw,
                                                      nullptr, rows, 2 * Cc, Cc);
  // rope + per-head layouts (q carries the 1/sqrt(D)=0.125 softmax scale)
  int ropeN = Bc * Hc * TQc * (Dc / 2);
  rope_kernel<<<(ropeN + 255) / 256, 256, 0, stream>>>(qraw, Cc, y_t, freqs, qh,
                                                       TQc, 0.125f);
  rope_kernel<<<(ropeN + 255) / 256, 256, 0, stream>>>(kvraw, 2 * Cc, x_t, freqs,
                                                       kh, TKc, 1.0f);
  int vN = Bc * TKc * Cc;
  cast_v_kernel<<<(vN + 255) / 256, 256, 0, stream>>>(kvraw, vt);
  // masked flash attention (4 waves / workgroup, one 16-row q-tile each)
  flash_attn_kernel<<<dim3(TQc / 64, Hc, Bc), 128, 0, stream>>>(qh, kh, vt, y_t,
                                                                x_t, mind, oh);
  // y1 = y + o @ Wproj^T  (y1 aliases qraw, already consumed)
  float* y1 = qraw;
  gemm_wmma_kernel<EPI_RESID_F32><<<512, 256, 0, stream>>>(oh, wprojh, y, y1,
                                                           nullptr, rows, Cc, Cc);
  // MLP
  ln_cast_kernel<<<rows, 256, 0, stream>>>(y1, ln2w, qnh, Cc);
  gemm_wmma_kernel<EPI_GELU_F16><<<512, 256, 0, stream>>>(qnh, wfch, nullptr,
                                                          nullptr, knh, rows, Cc, Cc);
  gemm_wmma_kernel<EPI_RESID_F32><<<512, 256, 0, stream>>>(knh, wmlph, y1,
                                                           (float*)d_out, nullptr,
                                                           rows, Cc, Cc);
  // second output: x passthrough
  hipMemcpyAsync((char*)d_out + (size_t)rows * Cc * sizeof(float), x,
                 (size_t)Bc * TKc * Cc * sizeof(float),
                 hipMemcpyDeviceToDevice, stream);
}
